// ExponentiatedQuadratic_59021440582478
// MI455X (gfx1250) — matrix operand order repaired and checked by reading, not compiled, not run on hardware
//
#include <hip/hip_runtime.h>

typedef __attribute__((ext_vector_type(16))) _Float16 v16h;
typedef __attribute__((ext_vector_type(8)))  _Float16 v8h;
typedef __attribute__((ext_vector_type(8)))  float    v8f;

#define NN 4096
#define MM 4096
#define PP 32

// ---------------------------------------------------------------------------
// Stage 1: per-row preprocessing.
//   xs = x / rho  (rho = exp(log_rho))
//   split each f32 value into f16 hi + f16 lo (hi = f16(v), lo = f16(v - hi))
//   row norm ||xs||^2 in f32 via wave butterfly reduce (wave32).
// One wave (32 lanes) per row; lane == column index (P == 32).
// ---------------------------------------------------------------------------
__global__ void __launch_bounds__(256)
eq_prep(const float* __restrict__ x1, const float* __restrict__ x2,
        const float* __restrict__ log_rho,
        _Float16* __restrict__ xh, _Float16* __restrict__ xl,
        _Float16* __restrict__ yh, _Float16* __restrict__ yl,
        float* __restrict__ xn, float* __restrict__ yn) {
  const int wid  = threadIdx.x >> 5;
  const int lane = threadIdx.x & 31;
  const int row  = blockIdx.x * 8 + wid;          // 0 .. NN+MM-1
  const float rinv = __expf(-log_rho[0]);         // 1/rho

  const float* src;
  _Float16 *dh, *dl;
  float* dn;
  int r;
  if (row < NN) { src = x1; dh = xh; dl = xl; dn = xn; r = row; }
  else          { src = x2; dh = yh; dl = yl; dn = yn; r = row - NN; }

  const float v = src[r * PP + lane] * rinv;
  const _Float16 h = (_Float16)v;
  const _Float16 l = (_Float16)(v - (float)h);
  dh[r * PP + lane] = h;
  dl[r * PP + lane] = l;

  float s = v * v;
#pragma unroll
  for (int off = 16; off >= 1; off >>= 1) s += __shfl_xor(s, off, 32);
  if (lane == 0) dn[r] = s;
}

// ---------------------------------------------------------------------------
// Stage 2: tiled WMMA GEMM + RBF epilogue.
// One wave per 16x16 output tile; 8 waves/block -> 32(M) x 64(N) region.
// dot(xs_i, ys_j) via split products: Ah*Bh + Ah*Bl + Al*Bh (f32 accum).
// Fragment layouts (wave32); A/B use this code's contiguous K order, not the ISA table order:
//   A (16x32 f16): row = lane%16; halfs 0..15 -> K = 16*(lane/16)+h
//   B (32x16 f16): col = lane%16; halfs 0..15 -> K = 16*(lane/16)+h
//   C/D (16x16 f32): reg r -> M = r + 8*(lane/16), N = lane%16
// ---------------------------------------------------------------------------
__global__ void __launch_bounds__(256)
eq_gemm(const _Float16* __restrict__ xh, const _Float16* __restrict__ xl,
        const _Float16* __restrict__ yh, const _Float16* __restrict__ yl,
        const float* __restrict__ xn, const float* __restrict__ yn,
        const float* __restrict__ log_sigma, float* __restrict__ out) {
  const int wid  = threadIdx.x >> 5;
  const int lane = threadIdx.x & 31;
  const int hsel = lane >> 4;      // 0 or 1: which half of the wave
  const int lr   = lane & 15;

  const int tile_m = blockIdx.y * 2 + (wid >> 2);
  const int tile_n = blockIdx.x * 4 + (wid & 3);
  const int m_base = tile_m * 16;  // rows of x1
  const int n_base = tile_n * 16;  // rows of x2 (columns of K)

  // ---- A fragments (x1 side), two 16B loads per fragment ----
  const int row_a = m_base + lr;
  const _Float16* pah = xh + (size_t)row_a * PP;
  const _Float16* pal = xl + (size_t)row_a * PP;
  const v8h ah0 = *(const v8h*)(pah + 16 * hsel);
  const v8h ah1 = *(const v8h*)(pah + 8 + 16 * hsel);
  const v8h al0 = *(const v8h*)(pal + 16 * hsel);
  const v8h al1 = *(const v8h*)(pal + 8 + 16 * hsel);
  v16h Ah, Al;
#pragma unroll
  for (int i = 0; i < 8; ++i) {
    Ah[i] = ah0[i]; Ah[i + 8] = ah1[i];
    Al[i] = al0[i]; Al[i + 8] = al1[i];
  }

  // ---- B fragments (x2 side), 16 contiguous K halfs per lane ----
  const int col_b = n_base + lr;
  const v16h Bh = *(const v16h*)(yh + (size_t)col_b * PP + 16 * hsel);
  const v16h Bl = *(const v16h*)(yl + (size_t)col_b * PP + 16 * hsel);

  // ---- split-product WMMA accumulation: hi*hi + hi*lo + lo*hi ----
  v8f acc = {};
  acc = __builtin_amdgcn_wmma_f32_16x16x32_f16(false, Ah, false, Bh,
                                               (short)0, acc, false, false);
  acc = __builtin_amdgcn_wmma_f32_16x16x32_f16(false, Ah, false, Bl,
                                               (short)0, acc, false, false);
  acc = __builtin_amdgcn_wmma_f32_16x16x32_f16(false, Al, false, Bh,
                                               (short)0, acc, false, false);

  // ---- epilogue: K = sigma^2 * exp(-0.5 d) = exp(2*log_sigma - 0.5 d) ----
  const float two_ls = 2.0f * log_sigma[0];
  const float ync = yn[col_b];
#pragma unroll
  for (int r = 0; r < 8; ++r) {
    const int row = m_base + r + 8 * hsel;
    float d = xn[row] + ync - 2.0f * acc[r];
    d = fmaxf(d, 0.0f);
    out[(size_t)row * MM + col_b] = __expf(two_ls - 0.5f * d);
  }
}

// ---------------------------------------------------------------------------
// Launch wrapper. Workspace layout (1,081,344 bytes):
//   [0)              xh : NN*PP f16
//   [+256KB)         xl : NN*PP f16
//   [+512KB)         yh : MM*PP f16
//   [+768KB)         yl : MM*PP f16
//   [+1MB)           xn : NN f32
//   [+1MB+16KB)      yn : MM f32
// ---------------------------------------------------------------------------
extern "C" void kernel_launch(void* const* d_in, const int* in_sizes, int n_in,
                              void* d_out, int out_size, void* d_ws, size_t ws_size,
                              hipStream_t stream) {
  const float* x1        = (const float*)d_in[0];
  const float* x2        = (const float*)d_in[1];
  const float* log_rho   = (const float*)d_in[2];
  const float* log_sigma = (const float*)d_in[3];
  float* out = (float*)d_out;

  _Float16* xh = (_Float16*)d_ws;
  _Float16* xl = xh + (size_t)NN * PP;
  _Float16* yh = xl + (size_t)NN * PP;
  _Float16* yl = yh + (size_t)MM * PP;
  float*    xn = (float*)(yl + (size_t)MM * PP);
  float*    yn = xn + NN;

  // Stage 1: (NN+MM) rows, 8 waves (rows) per 256-thread block.
  eq_prep<<<(NN + MM) / 8, 256, 0, stream>>>(x1, x2, log_rho,
                                             xh, xl, yh, yl, xn, yn);

  // Stage 2: 256x256 tiles of 16x16; block covers 2(M) x 4(N) tiles.
  dim3 grid(MM / 16 / 4, NN / 16 / 2);  // (64, 128)
  eq_gemm<<<grid, 256, 0, stream>>>(xh, xl, yh, yl, xn, yn, log_sigma, out);
}
